// ModernBertDecoderAttention_47562467836187
// MI455X (gfx1250) — matrix operand order repaired and checked by reading, not compiled, not run on hardware
//
#include <hip/hip_runtime.h>
#include <hip/hip_bf16.h>

// ---------------------------------------------------------------------------
// ModernBERT decoder attention (sliding window 64) for gfx1250 / MI455X.
// All matmuls via v_wmma_f32_16x16x32_f16 (f16 operands, fp32 accumulate).
// ---------------------------------------------------------------------------

typedef _Float16 h16;
typedef __attribute__((ext_vector_type(16))) _Float16 v16h;
typedef __attribute__((ext_vector_type(8)))  _Float16 v8h;
typedef __attribute__((ext_vector_type(8)))  float    v8f;

#define SEQ   4096
#define HDIM  768
#define NHEAD 12
#define HD    64
#define WND   64

__device__ __forceinline__ v8f vzero() {
    v8f z;
    #pragma unroll
    for (int i = 0; i < 8; ++i) z[i] = 0.0f;
    return z;
}

__device__ __forceinline__ v16h vcat(v8h a, v8h b) {
    return __builtin_shufflevector(a, b, 0,1,2,3,4,5,6,7,8,9,10,11,12,13,14,15);
}

__device__ __forceinline__ v8f wmma_f16(v16h a, v16h b, v8f c) {
    // emits v_wmma_f32_16x16x32_f16
    return __builtin_amdgcn_wmma_f32_16x16x32_f16(
        /*neg_a=*/false, a, /*neg_b=*/false, b,
        /*c_mod=*/(short)0, c, /*reuse_a=*/false, /*reuse_b=*/false);
}

// Build an A-fragment (16x32 f16, M x K) from a row-major fp32 row pointer.
// Lane layout: row m = lane%16; K halves [c, c+8) and [c+16, c+24), c = 8*(lane>=16).
__device__ __forceinline__ v16h a_frag_f32(const float* ap, int cA) {
    float4 f0 = *(const float4*)(ap + cA);
    float4 f1 = *(const float4*)(ap + cA + 4);
    float4 f2 = *(const float4*)(ap + cA + 16);
    float4 f3 = *(const float4*)(ap + cA + 20);
    v16h a;
    a[0]=(h16)f0.x; a[1]=(h16)f0.y; a[2]=(h16)f0.z; a[3]=(h16)f0.w;
    a[4]=(h16)f1.x; a[5]=(h16)f1.y; a[6]=(h16)f1.z; a[7]=(h16)f1.w;
    a[8]=(h16)f2.x; a[9]=(h16)f2.y; a[10]=(h16)f2.z; a[11]=(h16)f2.w;
    a[12]=(h16)f3.x; a[13]=(h16)f3.y; a[14]=(h16)f3.z; a[15]=(h16)f3.w;
    return a;
}

// ---------------------------------------------------------------------------
// Kernel 1: fused QKV projection + RoPE.
//   grid = (S/256, 2304/64), block = 256 (8 waves).
//   Each wave computes a 32x64 output tile (two 16-row tiles, one head's dims).
//   Outputs: Qh, Kh as f16 [head][s][64]; V transposed Vt f16 [head][64][s].
// ---------------------------------------------------------------------------
__global__ __launch_bounds__(256)
void qkv_rope_kernel(const float* __restrict__ hidden,
                     const float* __restrict__ wqkv,
                     const float* __restrict__ cosT,
                     const float* __restrict__ sinT,
                     h16* __restrict__ Qh, h16* __restrict__ Kh,
                     h16* __restrict__ Vt)
{
    alignas(16) __shared__ h16 Bs[64 * 32];   // W tile: row n (out col), K contiguous

    const int tid  = threadIdx.x;
    const int lane = tid & 31;
    const int wave = tid >> 5;
    const int ml   = lane & 15;
    const int hi   = lane >> 4;               // 0 or 1
    const int s0   = blockIdx.x * 256;
    const int g    = blockIdx.y;              // column group of 64
    const int n0   = g * 64;

    const int cA = hi << 3;                   // A-frag K base: 0 or 8
    const int cB = hi << 4;                   // 0 or 16; unused here (B-frag uses cA)
    const int srow = s0 + (wave << 5) + ml;   // rt=0 row; rt=1 adds 16

    v8f acc[2][4];
    #pragma unroll
    for (int rt = 0; rt < 2; ++rt)
        #pragma unroll
        for (int i = 0; i < 4; ++i) acc[rt][i] = vzero();

    for (int k0 = 0; k0 < HDIM; k0 += 32) {
        // stage W tile -> LDS (f32 -> f16): 64 rows x 32 K, 8 elems/thread
        {
            const int n = tid >> 2, q = tid & 3;
            const float* src = wqkv + (size_t)(n0 + n) * HDIM + k0 + q * 8;
            h16* dst = Bs + n * 32 + q * 8;
            #pragma unroll
            for (int i = 0; i < 8; ++i) dst[i] = (h16)src[i];
        }
        __syncthreads();

        // two A-frags direct from global fp32 (rows srow and srow+16)
        v16h a0 = a_frag_f32(hidden + (size_t)srow * HDIM + k0, cA);
        v16h a1 = a_frag_f32(hidden + (size_t)(srow + 16) * HDIM + k0, cA);

        #pragma unroll
        for (int nt = 0; nt < 4; ++nt) {
            const h16* br = Bs + ((nt << 4) + ml) * 32 + cA;
            v16h b = vcat(*(const v8h*)br, *(const v8h*)(br + 16));
            acc[0][nt] = wmma_f16(a0, b, acc[0][nt]);
            acc[1][nt] = wmma_f16(a1, b, acc[1][nt]);
        }
        __syncthreads();
    }

    // epilogue: C layout -> lane holds col n = n0 + 16*nt + ml, rows m = r + 8*hi
    const int m_base = hi << 3;
    const int which  = g / NHEAD;             // 0=Q 1=K 2=V
    const int head   = g % NHEAD;

    #pragma unroll
    for (int rt = 0; rt < 2; ++rt) {
        const int sbase = s0 + (wave << 5) + (rt << 4) + m_base;
        if (which < 2) {
            // RoPE: pair tiles (0,2),(1,3); cos[s][d] == cos[s][d+32] for d<32
            #pragma unroll
            for (int p = 0; p < 2; ++p) {
                const int da = (p << 4) + ml;              // 0..31
                #pragma unroll
                for (int r = 0; r < 8; ++r) {
                    const int s = sbase + r;
                    const float c  = cosT[s * HD + da];
                    const float sn = sinT[s * HD + da];
                    const float x = acc[rt][p][r], y = acc[rt][p + 2][r];
                    acc[rt][p][r]     = x * c - y * sn;
                    acc[rt][p + 2][r] = y * c + x * sn;
                }
            }
            h16* dst = (which == 0) ? Qh : Kh;
            #pragma unroll
            for (int nt = 0; nt < 4; ++nt) {
                const int d = (nt << 4) + ml;
                #pragma unroll
                for (int r = 0; r < 8; ++r)
                    dst[((size_t)head * SEQ + sbase + r) * HD + d] =
                        (h16)acc[rt][nt][r];
            }
        } else {
            #pragma unroll
            for (int nt = 0; nt < 4; ++nt) {
                const int d = (nt << 4) + ml;
                #pragma unroll
                for (int r = 0; r < 8; ++r)
                    Vt[((size_t)head * HD + d) * SEQ + sbase + r] =
                        (h16)acc[rt][nt][r];
            }
        }
    }
}

// ---------------------------------------------------------------------------
// Kernel 2: sliding-window attention, one wave per (16-query tile, head).
//   grid = (S/16, NHEAD), block = 32.
// ---------------------------------------------------------------------------
__global__ __launch_bounds__(32)
void attn_kernel(const h16* __restrict__ Qh, const h16* __restrict__ Kh,
                 const h16* __restrict__ Vt, h16* __restrict__ AO)
{
    alignas(16) __shared__ float Sf[16 * 96];
    alignas(16) __shared__ h16   Pf[16 * 96];

    const int lane = threadIdx.x;
    const int ml   = lane & 15;
    const int hi   = lane >> 4;
    const int s0   = blockIdx.x * 16;
    const int head = blockIdx.y;

    const int qt     = s0 >> 4;
    const int nk     = (qt < 4 ? qt : 4) + 1;      // key tiles (1..5)
    const int kstart = s0 - (nk - 1) * 16;

    const int cA = hi << 3;
    const int cB = hi << 4;

    // Q A-frags for both K-dim chunks (head dim 0..31 and 32..63)
    const h16* qrow = Qh + ((size_t)head * SEQ + s0 + ml) * HD;
    v16h aq0 = vcat(*(const v8h*)(qrow + cA),      *(const v8h*)(qrow + cA + 16));
    v16h aq1 = vcat(*(const v8h*)(qrow + 32 + cA), *(const v8h*)(qrow + 32 + cA + 16));

    // scores = Q K^T, one 16x16 tile per key tile
    for (int t = 0; t < nk; ++t) {
        const h16* krow = Kh + ((size_t)head * SEQ + kstart + t * 16 + ml) * HD;
        v16h b0 = vcat(*(const v8h*)(krow + cA),      *(const v8h*)(krow + cA + 16));
        v16h b1 = vcat(*(const v8h*)(krow + 32 + cA), *(const v8h*)(krow + 32 + cA + 16));
        v8f c = vzero();
        c = wmma_f16(aq0, b0, c);
        c = wmma_f16(aq1, b1, c);
        #pragma unroll
        for (int r = 0; r < 8; ++r)
            Sf[(r + (hi << 3)) * 96 + t * 16 + ml] = c[r];
    }
    __syncthreads();

    // fp32 softmax: one lane per query row; analytic sliding-causal mask
    if (lane < 16) {
        const int m = lane, i = s0 + m;
        int lo = i - WND - kstart; if (lo < 0) lo = 0;
        const int hc = i - kstart;                   // inclusive upper col
        float mx = -3.4e38f;
        for (int cc = lo; cc <= hc; ++cc) {
            const float v = Sf[m * 96 + cc] * 0.125f;
            if (v > mx) mx = v;
        }
        float sum = 0.0f;
        for (int cc = 0; cc < 96; ++cc) {
            float p = 0.0f;
            if (cc >= lo && cc <= hc) {
                p = __expf(Sf[m * 96 + cc] * 0.125f - mx);
                sum += p;
            }
            Sf[m * 96 + cc] = p;                     // reuse Sf for unnormalized p
        }
        const float inv = 1.0f / sum;
        for (int cc = 0; cc < 96; ++cc)
            Pf[m * 96 + cc] = (h16)(Sf[m * 96 + cc] * inv);
    }
    __syncthreads();

    // out = P @ V  (K dim = keys, padded with zero probs to chunk of 32)
    const int nch = (nk + 1) >> 1;
    v8f o[4];
    #pragma unroll
    for (int i = 0; i < 4; ++i) o[i] = vzero();

    for (int ch = 0; ch < nch; ++ch) {
        const h16* prow = Pf + ml * 96 + ch * 32;
        v16h ap = vcat(*(const v8h*)(prow + cB), *(const v8h*)(prow + cB + 8));
        int kb = kstart + ch * 32 + cB;
        if (kb > SEQ - 16) kb = SEQ - 16;            // clamp: P is zero there anyway
        #pragma unroll
        for (int nt = 0; nt < 4; ++nt) {
            const h16* vrow = Vt + ((size_t)head * HD + nt * 16 + ml) * SEQ + kb;
            v16h bv = vcat(*(const v8h*)vrow, *(const v8h*)(vrow + 8));
            o[nt] = wmma_f16(ap, bv, o[nt]);
        }
    }

    // store f16 attention output as [s][head*64 + d] for the out-projection
    #pragma unroll
    for (int nt = 0; nt < 4; ++nt) {
        const int d = head * HD + nt * 16 + ml;
        #pragma unroll
        for (int r = 0; r < 8; ++r) {
            const int s = s0 + (hi << 3) + r;
            AO[(size_t)s * HDIM + d] = (h16)o[nt][r];
        }
    }
}

// ---------------------------------------------------------------------------
// Kernel 3: output projection out = AO @ Wo^T (fp32 output).
//   grid = (S/256, 768/64), block = 256 (8 waves), 32x64 tile per wave.
// ---------------------------------------------------------------------------
__global__ __launch_bounds__(256)
void oproj_kernel(const h16* __restrict__ AO,
                  const float* __restrict__ wo,
                  float* __restrict__ out)
{
    alignas(16) __shared__ h16 Bs[64 * 32];

    const int tid  = threadIdx.x;
    const int lane = tid & 31;
    const int wave = tid >> 5;
    const int ml   = lane & 15;
    const int hi   = lane >> 4;
    const int s0   = blockIdx.x * 256;
    const int n0   = blockIdx.y * 64;

    const int cA = hi << 3;
    const int cB = hi << 4;
    const int srow = s0 + (wave << 5) + ml;

    v8f acc[2][4];
    #pragma unroll
    for (int rt = 0; rt < 2; ++rt)
        #pragma unroll
        for (int i = 0; i < 4; ++i) acc[rt][i] = vzero();

    for (int k0 = 0; k0 < HDIM; k0 += 32) {
        {
            const int n = tid >> 2, q = tid & 3;
            const float* src = wo + (size_t)(n0 + n) * HDIM + k0 + q * 8;
            h16* dst = Bs + n * 32 + q * 8;
            #pragma unroll
            for (int i = 0; i < 8; ++i) dst[i] = (h16)src[i];
        }
        __syncthreads();

        const h16* ap0 = AO + (size_t)srow * HDIM + k0;
        const h16* ap1 = AO + (size_t)(srow + 16) * HDIM + k0;
        v16h a0 = vcat(*(const v8h*)(ap0 + cA), *(const v8h*)(ap0 + cA + 16));
        v16h a1 = vcat(*(const v8h*)(ap1 + cA), *(const v8h*)(ap1 + cA + 16));

        #pragma unroll
        for (int nt = 0; nt < 4; ++nt) {
            const h16* br = Bs + ((nt << 4) + ml) * 32 + cA;
            v16h b = vcat(*(const v8h*)br, *(const v8h*)(br + 16));
            acc[0][nt] = wmma_f16(a0, b, acc[0][nt]);
            acc[1][nt] = wmma_f16(a1, b, acc[1][nt]);
        }
        __syncthreads();
    }

    const int m_base = hi << 3;
    #pragma unroll
    for (int rt = 0; rt < 2; ++rt) {
        const int sbase = s0 + (wave << 5) + (rt << 4) + m_base;
        #pragma unroll
        for (int nt = 0; nt < 4; ++nt) {
            const int n = n0 + (nt << 4) + ml;
            #pragma unroll
            for (int r = 0; r < 8; ++r)
                out[(size_t)(sbase + r) * HDIM + n] = acc[rt][nt][r];
        }
    }
}

// ---------------------------------------------------------------------------
extern "C" void kernel_launch(void* const* d_in, const int* in_sizes, int n_in,
                              void* d_out, int out_size, void* d_ws, size_t ws_size,
                              hipStream_t stream) {
    const float* hidden = (const float*)d_in[0];   // [4096,768]
    const float* wqkv   = (const float*)d_in[1];   // [2304,768]
    const float* wo     = (const float*)d_in[2];   // [768,768]
    const float* cosT   = (const float*)d_in[3];   // [4096,64]
    const float* sinT   = (const float*)d_in[4];   // [4096,64]
    // d_in[5] = attention_mask: recomputed analytically, never read.
    float* out = (float*)d_out;

    char* ws = (char*)d_ws;
    const size_t SZ = (size_t)NHEAD * SEQ * HD * sizeof(h16);   // 6 MB each
    h16* Qh = (h16*)(ws);
    h16* Kh = (h16*)(ws + SZ);
    h16* Vt = (h16*)(ws + 2 * SZ);
    h16* AO = (h16*)(ws + 3 * SZ);

    qkv_rope_kernel<<<dim3(SEQ / 256, 36), dim3(256), 0, stream>>>(
        hidden, wqkv, cosT, sinT, Qh, Kh, Vt);
    attn_kernel<<<dim3(SEQ / 16, NHEAD), dim3(32), 0, stream>>>(
        Qh, Kh, Vt, AO);
    oproj_kernel<<<dim3(SEQ / 256, HDIM / 64), dim3(256), 0, stream>>>(
        AO, wo, out);
}
